// ReactionVariabilitySystem_84877143703993
// MI455X (gfx1250) — matrix operand order repaired and checked by reading, not compiled, not run on hardware
//
#include <hip/hip_runtime.h>
#include <math.h>

#define BATCH 256
#define SEQ   4096
#define HDIM  4096
#define VOCAB 32000

typedef __attribute__((ext_vector_type(16))) __bf16 v16bf;
typedef __attribute__((ext_vector_type(2)))  __bf16 v2bf;
typedef __attribute__((ext_vector_type(8)))  float  v8f;
typedef __attribute__((__vector_size__(16))) int    v4i;   // int4, matches builtin proto

#if __has_builtin(__builtin_amdgcn_global_load_async_to_lds_b128)
#define ASYNC_LDS 1
#endif

// ---------------------------------------------------------------------------
// Kernel 1: fused GEMM (hidden@W^T and prev@W^T) + cosine-similarity partials.
// Grid: (HDIM/128, BATCH/16). Block: 256 threads = 8 waves.
// LDS tiles are stored in WMMA-fragment order so every operand fragment is a
// single 32-byte contiguous load (2x ds_load_b128): A is stored [m][k] and
// read in contiguous k order (k = 16*half + s); B is stored [n][k].
// Row stride 72 bf16 (144B) keeps b128 fragment reads bank-conflict-free and
// 16B-aligned. Each wave runs TWO wmma chains (hidden, prev) sharing one
// B-fragment; per-row sums reduced by xor-shuffles + 3 atomics.
// ---------------------------------------------------------------------------
__global__ __launch_bounds__(256) void sim_gemm_kernel(
    const float* __restrict__ hidden, const float* __restrict__ prev,
    const float* __restrict__ W,
    float* __restrict__ s11, float* __restrict__ s22, float* __restrict__ s12)
{
    __shared__ __align__(16) __bf16 lA1[16 * 72];
    __shared__ __align__(16) __bf16 lA2[16 * 72];
    __shared__ __align__(16) __bf16 lB [128 * 72];

    const int tid  = threadIdx.x;
    const int wave = tid >> 5;
    const int lane = tid & 31;
    const int half = lane >> 4;          // 0: lanes 0-15, 1: lanes 16-31
    const int l16  = lane & 15;
    const int m0   = blockIdx.y * 16;
    const int n0   = blockIdx.x * 128;

    // A staging: thread -> 2 consecutive elems: row ar, cols (ac, ac+1).
    // Fragment map: contiguous k order, k -> (hw=k>>4, ew=k&15), matching B
    const int ar   = tid >> 4;
    const int ac   = (tid << 1) & 31;
    const int akk  = ac & 15;
    const int aoff = ar * 72 + ac;

    // B staging: thread -> 16 consecutive k for row n: n = tid>>1, c0 = (tid&1)*16
    const int bn   = tid >> 1;
    const int bc   = (tid & 1) << 4;
    const int boff = bn * 72 + bc;

    // Fragment read offsets (32B contiguous per lane)
    const int a_rd = l16 * 72 + half * 16;
    const int b_rd = (wave * 16 + l16) * 72 + half * 16;

    v8f c1 = {};
    v8f c2 = {};

    for (int k0 = 0; k0 < HDIM; k0 += 32) {
        __syncthreads();
        {
            float2 f1 = *(const float2*)&hidden[(size_t)(m0 + ar) * HDIM + k0 + ac];
            float2 f2 = *(const float2*)&prev  [(size_t)(m0 + ar) * HDIM + k0 + ac];
            v2bf p1; p1[0] = (__bf16)f1.x; p1[1] = (__bf16)f1.y;
            v2bf p2; p2[0] = (__bf16)f2.x; p2[1] = (__bf16)f2.y;
            *(v2bf*)&lA1[aoff] = p1;
            *(v2bf*)&lA2[aoff] = p2;

            const float* wp = &W[(size_t)(n0 + bn) * HDIM + k0 + bc];
            v16bf wb;
#pragma unroll
            for (int i = 0; i < 16; ++i) wb[i] = (__bf16)wp[i];
            *(v16bf*)&lB[boff] = wb;
        }
        if (k0 + 32 < HDIM)   // hint next W tile toward the caches
            __builtin_prefetch(&W[(size_t)(n0 + bn) * HDIM + k0 + 32 + bc], 0, 1);
        __syncthreads();

        v16bf a1 = *(const v16bf*)&lA1[a_rd];
        v16bf a2 = *(const v16bf*)&lA2[a_rd];
        v16bf bf = *(const v16bf*)&lB [b_rd];
        c1 = __builtin_amdgcn_wmma_f32_16x16x32_bf16(false, a1, false, bf,
                                                     (short)0, c1, false, false);
        c2 = __builtin_amdgcn_wmma_f32_16x16x32_bf16(false, a2, false, bf,
                                                     (short)0, c2, false, false);
    }

    // C/D layout: VGPR r -> M = r + 8*half, N = lane%16.
#pragma unroll
    for (int r = 0; r < 8; ++r) {
        float v11 = c1[r] * c1[r];
        float v22 = c2[r] * c2[r];
        float v12 = c1[r] * c2[r];
#pragma unroll
        for (int m = 1; m < 16; m <<= 1) {
            v11 += __shfl_xor(v11, m, 32);
            v22 += __shfl_xor(v22, m, 32);
            v12 += __shfl_xor(v12, m, 32);
        }
        if (l16 == 0) {
            int row = m0 + r + 8 * half;
            atomicAdd(&s11[row], v11);
            atomicAdd(&s22[row], v22);
            atomicAdd(&s12[row], v12);
        }
    }
}

// ---------------------------------------------------------------------------
// Kernel 2: 4-gram repetition score via LDS hash-set (one workgroup / batch).
// repeats = T - unique; each insert that finds its key present counts one.
// ---------------------------------------------------------------------------
__global__ __launch_bounds__(256) void ngram_kernel(
    const int* __restrict__ ids, float* __restrict__ rep)
{
    __shared__ unsigned long long table[8192];   // 64 KB, load factor ~0.5
    __shared__ unsigned int dups;
    const int b   = blockIdx.x;
    const int tid = threadIdx.x;
    for (int i = tid; i < 8192; i += 256) table[i] = 0ull;
    if (tid == 0) dups = 0;
    __syncthreads();

    const int T = SEQ - 4 + 1;
    unsigned int local = 0;
    for (int t = tid; t < T; t += 256) {
        const int* p = ids + (size_t)b * SEQ + t;
        unsigned long long hi = (unsigned long long)(unsigned)p[0] * 32000u + (unsigned)p[1];
        unsigned long long lo = (unsigned long long)(unsigned)p[2] * 32000u + (unsigned)p[3];
        unsigned long long key = ((hi << 30) | lo) + 1ull;   // hi,lo < 2^30; key != 0
        unsigned long long h = key * 0x9E3779B97F4A7C15ull;
        unsigned int slot = (unsigned int)(h >> 32) & 8191u;
        for (;;) {
            unsigned long long old = atomicCAS(&table[slot], 0ull, key);
            if (old == 0ull) break;               // inserted -> first occurrence
            if (old == key) { local++; break; }   // seen before -> repeat
            slot = (slot + 1u) & 8191u;           // linear probe
        }
    }
    atomicAdd(&dups, local);
    __syncthreads();
    if (tid == 0) rep[b] = (float)dups / (float)T;
}

// ---------------------------------------------------------------------------
// Kernel 3: per-batch token histogram + output.
// ids row (16KB raw int32) staged into LDS with GLOBAL_LOAD_ASYNC_TO_LDS_B128
// (ASYNCcnt path) when available; 4 passes of an 8000-bin LDS histogram.
// out[b,v] = logits[b,v] + bias[v] - freq[b,v] * (0.1 / (4096 + 1e-8))
// ---------------------------------------------------------------------------
#define FBINS 8000
__global__ __launch_bounds__(256) void freq_out_kernel(
    const float* __restrict__ logits, const int* __restrict__ ids,
    const float* __restrict__ bias, float* __restrict__ out)
{
    __shared__ unsigned int hist[FBINS];         // 32 KB
    __shared__ __align__(16) int sids[SEQ];      // 16 KB
    const int b   = blockIdx.x;
    const int tid = threadIdx.x;

#ifdef ASYNC_LDS
#pragma unroll
    for (int i = 0; i < 4; ++i) {
        int idx = (tid + i * 256) * 4;           // 4 ints = 16B per lane per op
        __builtin_amdgcn_global_load_async_to_lds_b128(
            (v4i*)(void*)(ids + (size_t)b * SEQ + idx),
            (v4i*)(void*)(sids + idx), 0, 0);
    }
#if __has_builtin(__builtin_amdgcn_s_wait_asynccnt)
    __builtin_amdgcn_s_wait_asynccnt(0);
#else
    asm volatile("s_wait_asynccnt 0x0" ::: "memory");
#endif
    __syncthreads();
#else
    for (int i = tid; i < SEQ; i += 256) sids[i] = ids[(size_t)b * SEQ + i];
    __syncthreads();
#endif

    const float scale = 0.1f / (4096.0f + 1e-8f);
    for (int p = 0; p < 4; ++p) {                // 4 * 8000 == VOCAB
        const int base = p * FBINS;
        for (int i = tid; i < FBINS; i += 256) hist[i] = 0u;
        __syncthreads();
        for (int t = tid; t < SEQ; t += 256) {
            int rel = sids[t] - base;
            if (rel >= 0 && rel < FBINS) atomicAdd(&hist[rel], 1u);
        }
        __syncthreads();
        for (int i = tid; i < FBINS; i += 256) {
            int v = base + i;
            out[(size_t)b * VOCAB + v] =
                logits[(size_t)b * VOCAB + v] + bias[v] - (float)hist[i] * scale;
        }
        __syncthreads();
    }
}

// ---------------------------------------------------------------------------
// Kernel 4: finalize temperature from reduction partials + rep score.
// ---------------------------------------------------------------------------
__global__ void temp_kernel(const float* __restrict__ s11,
                            const float* __restrict__ s22,
                            const float* __restrict__ s12,
                            const float* __restrict__ rep,
                            float* __restrict__ out_temp)
{
    int b = blockIdx.x * blockDim.x + threadIdx.x;
    if (b >= BATCH) return;
    float n1  = fmaxf(sqrtf(s11[b]), 1e-8f);
    float n2  = fmaxf(sqrtf(s22[b]), 1e-8f);
    float sim = s12[b] / (n1 * n2);
    float pen = fmaxf(sim, 0.0f) * 0.1f;               // PENALTY_WEIGHT
    out_temp[b] = 1.0f * (1.0f + (rep[b] + pen) * 0.5f) * 1.0f;
}

extern "C" void kernel_launch(void* const* d_in, const int* in_sizes, int n_in,
                              void* d_out, int out_size, void* d_ws, size_t ws_size,
                              hipStream_t stream)
{
    (void)in_sizes; (void)n_in; (void)out_size; (void)ws_size;
    const float* logits = (const float*)d_in[0];   // (B, V)
    const float* hidden = (const float*)d_in[1];   // (B, H)
    const float* prev   = (const float*)d_in[2];   // (B, H)
    const int*   ids    = (const int*)  d_in[3];   // (B, S)
    const float* W      = (const float*)d_in[4];   // (H, H)
    const float* bias   = (const float*)d_in[5];   // (V,)

    float* out  = (float*)d_out;                   // (B, V) then (B,) temperature
    float* temp = out + (size_t)BATCH * VOCAB;

    float* s11 = (float*)d_ws;                     // (B,)
    float* s22 = s11 + BATCH;                      // (B,)
    float* s12 = s22 + BATCH;                      // (B,)
    float* rep = s12 + BATCH;                      // (B,)

    (void)hipMemsetAsync(s11, 0, 3 * BATCH * sizeof(float), stream);

    dim3 ggrid(HDIM / 128, BATCH / 16);
    sim_gemm_kernel<<<ggrid, 256, 0, stream>>>(hidden, prev, W, s11, s22, s12);
    ngram_kernel<<<BATCH, 256, 0, stream>>>(ids, rep);
    freq_out_kernel<<<BATCH, 256, 0, stream>>>(logits, ids, bias, out);
    temp_kernel<<<1, 256, 0, stream>>>(s11, s22, s12, rep, temp);
}
